// GConv_4913442587201
// MI455X (gfx1250) — compile-verified, matrix operand order checked by reading, not run on hardware
//
#include <hip/hip_runtime.h>

typedef __attribute__((ext_vector_type(16))) __bf16 v16bf;
typedef __attribute__((ext_vector_type(8)))  float  v8f;

#define H_DIM 256   // hidden width (== in_sizes[3])

// ---------------------------------------------------------------------------
// Degree / normalization kernels: deg[i] = 1 (self loop) + #incoming edges,
// then dinv[i] = rsqrt(deg[i]).
// ---------------------------------------------------------------------------
__global__ void deg_init_kernel(float* __restrict__ deg, int N) {
    int i = blockIdx.x * blockDim.x + threadIdx.x;
    if (i < N) deg[i] = 1.0f;   // self-loop contribution
}

__global__ void deg_accum_kernel(const long long* __restrict__ dst,
                                 float* __restrict__ deg, int E) {
    int e = blockIdx.x * blockDim.x + threadIdx.x;
    if (e < E) atomicAdd(&deg[(int)dst[e]], 1.0f);
}

__global__ void dinv_kernel(float* __restrict__ deg, int N) {
    int i = blockIdx.x * blockDim.x + threadIdx.x;
    if (i < N) {
        float d = deg[i];
        deg[i] = (d > 0.0f) ? rsqrtf(d) : 0.0f;
    }
}

// ---------------------------------------------------------------------------
// Fragment-layout helpers for V_WMMA_F32_16X16X32_BF16 (16-bit 16x32 A / 32x16 B):
// lane = half*16 + (m|n), element e in lane's 16-value chunk maps to
// k = ((p&3)<<1) + ((p>>2)<<4) + (half<<3) + (e&1), p = e>>1.
// Inverse (from k): half=(k>>3)&1, p=((k>>1)&3)|(((k>>4)&1)<<2), e=(p<<1)|(k&1).
// ---------------------------------------------------------------------------
__device__ __forceinline__ int frag_half(int k) { return (k >> 3) & 1; }
__device__ __forceinline__ int frag_elem(int k) {
    int p = ((k >> 1) & 3) | (((k >> 4) & 1) << 2);
    return (p << 1) | (k & 1);
}

// ---------------------------------------------------------------------------
// T = X @ W  via V_WMMA_F32_16X16X32_BF16.
// Workgroup: 256 threads (8 waves). Tile: 128 rows x 64 cols.
// Wave w owns rows [w*16, w*16+16); 4 col-subtiles of 16.
// LDS holds tiles PRE-SWIZZLED into per-lane fragment order so the hot path
// is a single 32-byte LDS load (2x ds_load_b128) per fragment.
// Global loads for tile kb+32 are prefetched into registers while the WMMAs
// for tile kb execute.
// ---------------------------------------------------------------------------
__global__ __launch_bounds__(256)
void gemm_wmma_bf16_kernel(const float* __restrict__ X,
                           const float* __restrict__ W,
                           float* __restrict__ T,
                           int N, int K) {
    // [wave-tile 0..7][lane 0..31][16 bf16]  (each lane chunk 32B-aligned)
    __shared__ __bf16 lAf[8][32][16] __attribute__((aligned(32)));
    // [col-subtile 0..3][lane 0..31][16 bf16]
    __shared__ __bf16 lBf[4][32][16] __attribute__((aligned(32)));

    const int tid  = threadIdx.x;
    const int wave = tid >> 5;
    const int lane = tid & 31;
    const int rowBase = blockIdx.x * 128;
    const int colBase = blockIdx.y * 64;

    // ---- staging coordinates (fixed per thread) ----
    // A: thread covers row ar (0..127), cols acb..acb+15 of the 128x32 tile
    const int ar  = tid >> 1;
    const int acb = (tid & 1) * 16;
    const int aw  = ar >> 4;        // wave-tile index
    const int am  = ar & 15;        // m within subtile
    // B: thread covers k-row bk (0..31), cols bcb..bcb+7 of the 32x64 tile
    const int bk  = tid >> 3;
    const int bcb = (tid & 7) * 8;
    const int bj  = bcb >> 4;                 // col-subtile (constant per thread)
    const int be  = frag_elem(bk);            // element slot from k
    const int bhalf = frag_half(bk);

    v8f acc[4];
#pragma unroll
    for (int j = 0; j < 4; ++j)
#pragma unroll
        for (int r = 0; r < 8; ++r) acc[j][r] = 0.0f;

    // ---- register prefetch buffers ----
    float4 aR[4];
    float4 bR[2];
    const int arow = rowBase + ar;
    const bool aValid = (arow < N);

    auto load_tile = [&](int kb) {
        if (aValid) {
            const float4* ap = (const float4*)(X + (size_t)arow * K + kb + acb);
#pragma unroll
            for (int q = 0; q < 4; ++q) aR[q] = ap[q];
        } else {
#pragma unroll
            for (int q = 0; q < 4; ++q) aR[q] = make_float4(0.f, 0.f, 0.f, 0.f);
        }
        const float4* bp = (const float4*)(W + (size_t)(kb + bk) * H_DIM + colBase + bcb);
        bR[0] = bp[0];
        bR[1] = bp[1];
    };

    load_tile(0);

    for (int kb = 0; kb < K; kb += 32) {
        // ---- stage prefetched registers into swizzled LDS ----
        {
            // A: 8 packed-pair b32 stores (c even + c odd are adjacent elems)
            const float av[16] = { aR[0].x, aR[0].y, aR[0].z, aR[0].w,
                                   aR[1].x, aR[1].y, aR[1].z, aR[1].w,
                                   aR[2].x, aR[2].y, aR[2].z, aR[2].w,
                                   aR[3].x, aR[3].y, aR[3].z, aR[3].w };
#pragma unroll
            for (int q = 0; q < 8; ++q) {
                int c    = acb + 2 * q;
                int h    = frag_half(c);
                int e    = frag_elem(c);           // even
                int ln   = (h << 4) | am;
                union { __bf16 hh[2]; unsigned int u; } pk;
                pk.hh[0] = (__bf16)av[2 * q + 0];
                pk.hh[1] = (__bf16)av[2 * q + 1];
                *(unsigned int*)&lAf[aw][ln][e] = pk.u;
            }
            // B: 8 scattered b16 stores (lane varies with column)
            const float bv[8] = { bR[0].x, bR[0].y, bR[0].z, bR[0].w,
                                  bR[1].x, bR[1].y, bR[1].z, bR[1].w };
#pragma unroll
            for (int u = 0; u < 8; ++u) {
                int c  = bcb + u;
                int ln = (bhalf << 4) | (c & 15);
                lBf[bj][ln][be] = (__bf16)bv[u];
            }
        }
        __syncthreads();

        // ---- prefetch next K-tile while WMMAs run ----
        if (kb + 32 < K) load_tile(kb + 32);

        // ---- fragments: one 32B load each (2x ds_load_b128) ----
        v16bf afrag = *(const v16bf*)&lAf[wave][lane][0];
#pragma unroll
        for (int j = 0; j < 4; ++j) {
            v16bf bfrag = *(const v16bf*)&lBf[j][lane][0];
            acc[j] = __builtin_amdgcn_wmma_f32_16x16x32_bf16(
                false, afrag, false, bfrag, (short)0, acc[j], false, false);
        }
        __syncthreads();
    }

    // ---- store per 16x16 f32 C/D layout: VGPR r -> M = r + half*8, N = lane&15
    const int m    = lane & 15;
    const int half = lane >> 4;
#pragma unroll
    for (int j = 0; j < 4; ++j) {
#pragma unroll
        for (int r = 0; r < 8; ++r) {
            int row = rowBase + wave * 16 + half * 8 + r;
            int col = colBase + j * 16 + m;
            if (row < N) T[(size_t)row * H_DIM + col] = acc[j][r];
        }
    }
}

// ---------------------------------------------------------------------------
// out[i] = T[i] * dinv[i]^2   (self-loop term; also initializes out buffer)
// ---------------------------------------------------------------------------
__global__ void self_loop_init_kernel(const float* __restrict__ T,
                                      const float* __restrict__ dinv,
                                      float* __restrict__ out) {
    int row = blockIdx.x;
    int f   = threadIdx.x;
    float s = dinv[row];
    out[(size_t)row * H_DIM + f] = T[(size_t)row * H_DIM + f] * s * s;
}

// ---------------------------------------------------------------------------
// For each edge e: out[dst] += T[src] * (dinv[src]*dinv[dst])
// One block of 256 threads = 256 features = one edge. out fits in L2 (102 MB
// < 192 MB) so the f32 atomics resolve at L2 bandwidth, not HBM.
// ---------------------------------------------------------------------------
__global__ __launch_bounds__(256)
void edge_scatter_kernel(const float* __restrict__ T,
                         const long long* __restrict__ src,
                         const long long* __restrict__ dst,
                         const float* __restrict__ dinv,
                         float* __restrict__ out) {
    int e = blockIdx.x;
    int s = (int)src[e];
    int d = (int)dst[e];
    if (threadIdx.x == 0) {
        __builtin_prefetch((const void*)(T + (size_t)s * H_DIM), 0, 0);
    }
    float nm = dinv[s] * dinv[d];
    int f = threadIdx.x;
    float v = T[(size_t)s * H_DIM + f] * nm;
    atomicAdd(&out[(size_t)d * H_DIM + f], v);
}

// ---------------------------------------------------------------------------
// out = relu(out + b)
// ---------------------------------------------------------------------------
__global__ void bias_relu_kernel(float* __restrict__ out,
                                 const float* __restrict__ b) {
    int row = blockIdx.x;
    int f   = threadIdx.x;
    size_t idx = (size_t)row * H_DIM + f;
    out[idx] = fmaxf(out[idx] + b[f], 0.0f);
}

// ---------------------------------------------------------------------------
extern "C" void kernel_launch(void* const* d_in, const int* in_sizes, int n_in,
                              void* d_out, int out_size, void* d_ws, size_t ws_size,
                              hipStream_t stream) {
    const float*     x  = (const float*)d_in[0];
    const long long* ei = (const long long*)d_in[1];
    const float*     W0 = (const float*)d_in[2];
    const float*     b0 = (const float*)d_in[3];
    const float*     W1 = (const float*)d_in[4];
    const float*     b1 = (const float*)d_in[5];
    const float*     W2 = (const float*)d_in[6];
    const float*     b2 = (const float*)d_in[7];

    const int H = in_sizes[3];          // 256
    const int D = in_sizes[2] / H;      // 128
    const int N = in_sizes[0] / D;      // 100000
    const int E = in_sizes[1] / 2;      // 640000

    const long long* src = ei;
    const long long* dst = ei + E;

    // Workspace layout: dinv (N floats) | T (N*H floats)
    char*  ws   = (char*)d_ws;
    float* dinv = (float*)ws;
    size_t off  = ((size_t)N * sizeof(float) + 255) & ~(size_t)255;
    float* T    = (float*)(ws + off);
    float* out  = (float*)d_out;

    // --- normalization ---
    deg_init_kernel <<<(N + 255) / 256, 256, 0, stream>>>(dinv, N);
    deg_accum_kernel<<<(E + 255) / 256, 256, 0, stream>>>(dst, dinv, E);
    dinv_kernel     <<<(N + 255) / 256, 256, 0, stream>>>(dinv, N);

    dim3 gemmGrid((N + 127) / 128, H / 64);

    // --- layer 0: h = relu(Â (x W0) + b0) ---
    gemm_wmma_bf16_kernel<<<gemmGrid, 256, 0, stream>>>(x, W0, T, N, D);
    self_loop_init_kernel<<<N, H, 0, stream>>>(T, dinv, out);
    edge_scatter_kernel  <<<E, H, 0, stream>>>(T, src, dst, dinv, out);
    bias_relu_kernel     <<<N, H, 0, stream>>>(out, b0);

    // --- layer 1 ---
    gemm_wmma_bf16_kernel<<<gemmGrid, 256, 0, stream>>>(out, W1, T, N, H);
    self_loop_init_kernel<<<N, H, 0, stream>>>(T, dinv, out);
    edge_scatter_kernel  <<<E, H, 0, stream>>>(T, src, dst, dinv, out);
    bias_relu_kernel     <<<N, H, 0, stream>>>(out, b1);

    // --- layer 2 ---
    gemm_wmma_bf16_kernel<<<gemmGrid, 256, 0, stream>>>(out, W2, T, N, H);
    self_loop_init_kernel<<<N, H, 0, stream>>>(T, dinv, out);
    edge_scatter_kernel  <<<E, H, 0, stream>>>(T, src, dst, dinv, out);
    bias_relu_kernel     <<<N, H, 0, stream>>>(out, b2);
}
